// c_Attention_43361989820757
// MI455X (gfx1250) — compile-verified, matrix operand order checked by reading, not run on hardware
//
#include <hip/hip_runtime.h>

// ---------------------------------------------------------------------------
// CDNA5 (gfx1250) fused masked-MHA + contrastive loss.
// GEMM stages: v_wmma_f32_16x16x32_bf16 (wave32).
// Async staging: global_load_async_to_lds_b128 + s_wait_asynccnt.
// Streaming outputs: non-temporal stores (preserve 192MB L2 for K/V/mask).
// ---------------------------------------------------------------------------

typedef __bf16 bf16_t;
typedef __attribute__((ext_vector_type(16))) __bf16 v16bf;
typedef __attribute__((ext_vector_type(8)))  __bf16 v8bf;
typedef __attribute__((ext_vector_type(8)))  float  v8f;

#define Bq 4
#define Nq 1024
#define DIMq 512
#define Hq 8
#define DHq 64
#define INNERq 512
#define SLD 1040   // padded f32 row stride for the score stripe (bank skew)

__device__ __forceinline__ v8f wmma_bf16(v16bf a, v16bf b, v8f c) {
  return __builtin_amdgcn_wmma_f32_16x16x32_bf16(
      /*neg_a=*/false, a, /*neg_b=*/false, b,
      /*c_mod=*/(short)0, c, /*reuse_a=*/false, /*reuse_b=*/false);
}

// Load a 16x32 bf16 fragment (A or B operand) for one lane from a row-major
// matrix. ISA layout: lane<16 holds K={0..7} and {16..23} of row (lane&15);
// lane>=16 holds K={8..15} and {24..31}. Each chunk = 8 bf16 = 16 bytes.
__device__ __forceinline__ v16bf load_frag(const bf16_t* __restrict__ base,
                                           int ld, int row, int kbase, int half) {
  const v8bf* p0 = (const v8bf*)(base + (size_t)row * ld + kbase + half * 8);
  const v8bf* p1 = (const v8bf*)(base + (size_t)row * ld + kbase + 16 + half * 8);
  v8bf lo = *p0;
  v8bf hi = *p1;
  v16bf r;
#pragma unroll
  for (int i = 0; i < 8; ++i) { r[i] = lo[i]; r[i + 8] = hi[i]; }
  return r;
}

// ---------------------------------------------------------------------------
// Elementwise f32 -> bf16 convert (x is read exactly once -> NT load)
// ---------------------------------------------------------------------------
__global__ void k_cvt(const float* __restrict__ in, bf16_t* __restrict__ out, int n) {
  int i = blockIdx.x * 256 + threadIdx.x;
  if (i < n) out[i] = (bf16_t)__builtin_nontemporal_load(in + i);
}

// Transposed convert: out[c][r] = in[r][c]
__global__ void k_cvt_t(const float* __restrict__ in, bf16_t* __restrict__ out,
                        int rows, int cols) {
  int i = blockIdx.x * 256 + threadIdx.x;
  if (i < rows * cols) {
    int r = i / cols, c = i % cols;
    out[(size_t)c * rows + r] = (bf16_t)in[i];
  }
}

// ---------------------------------------------------------------------------
// QKV projection: C[4096,512] = Xb[4096,512] @ W  (Wt given transposed [n][k])
// Grid: (64 row-blocks of 64) x (8 heads). 256 threads (8 waves).
// ---------------------------------------------------------------------------
__global__ void k_gemm_qkv(const bf16_t* __restrict__ X, const bf16_t* __restrict__ Wt,
                           const float* __restrict__ bias, bf16_t* __restrict__ out,
                           float* __restrict__ norms, int vtrans) {
  __shared__ float nsq[64];
  int tid = threadIdx.x;
  int lane = tid & 31, w = tid >> 5;
  int half = (lane >> 4) & 1, l16 = lane & 15;
  int rowblk = blockIdx.x * 64;
  int h = blockIdx.y;
  int rowtile = (w >> 1) * 16;   // 0,16,32,48
  int colpair = (w & 1) * 32;    // 0 or 32

  if (tid < 64) nsq[tid] = 0.0f;
  __syncthreads();

  v8f c0 = {}; v8f c1 = {};
  int arow = rowblk + rowtile + l16;
  int bcol0 = h * 64 + colpair;
  for (int k = 0; k < DIMq; k += 32) {
    v16bf a  = load_frag(X,  DIMq, arow,             k, half);
    v16bf b0 = load_frag(Wt, DIMq, bcol0 + l16,      k, half);
    v16bf b1 = load_frag(Wt, DIMq, bcol0 + 16 + l16, k, half);
    c0 = wmma_bf16(a, b0, c0);
    c1 = wmma_bf16(a, b1, c1);
  }

#pragma unroll
  for (int t = 0; t < 2; ++t) {
    v8f c = t ? c1 : c0;
    int colInBlk = colpair + t * 16 + l16;   // 0..63 (== dh)
    float bv = bias[h * 64 + colInBlk];
#pragma unroll
    for (int i = 0; i < 8; ++i) {
      int rInBlk = rowtile + i + half * 8;
      float v = c[i] + bv;
      int gm = rowblk + rInBlk;
      int b = gm >> 10, tok = gm & 1023;
      size_t addr;
      if (!vtrans)
        addr = ((size_t)((b * Hq + h) * Nq + tok)) * DHq + colInBlk;   // [b,h,tok,dh]
      else
        addr = ((size_t)((b * Hq + h) * DHq + colInBlk)) * Nq + tok;   // [b,h,dh,tok]
      out[addr] = (bf16_t)v;
      if (norms) atomicAdd(&nsq[rInBlk], v * v);                       // ds_add_f32
    }
  }
  if (norms) {
    __syncthreads();
    if (tid < 64) {
      int gm = rowblk + tid;
      int b = gm >> 10, tok = gm & 1023;
      norms[(size_t)(b * Hq + h) * Nq + tok] = sqrtf(nsq[tid]);
    }
  }
}

// ---------------------------------------------------------------------------
// Fused attention: per workgroup = 16 query rows of one (b,h).
//   async-stage Q tile -> LDS (ASYNCcnt)
//   Phase A: S = Q K^T (WMMA) into padded 16x1040 f32 LDS stripe
//   Phase B: DCL sums + masked softmax; single NT write of attn; bf16 P in LDS
//   Phase C: O = P @ V (WMMA, V pre-transposed), LDS f32 reduce, emit bf16
// ---------------------------------------------------------------------------
__global__ void k_attn(const bf16_t* __restrict__ Qb, const bf16_t* __restrict__ Kb,
                       const bf16_t* __restrict__ Vt, const float* __restrict__ qn_,
                       const float* __restrict__ kn_, const int* __restrict__ mask,
                       float* __restrict__ attn, bf16_t* __restrict__ inner,
                       float* __restrict__ accum) {
  extern __shared__ char smem[];
  // LDS carve (dynamic): S 16*1040*4 | P 16*1024*2 | Vacc 16*64*4 | red 4*256*4 | Qs 2KB
  const unsigned S_OFF  = 0;
  const unsigned P_OFF  = 16 * SLD * 4;
  const unsigned V_OFF  = P_OFF + 16 * 1024 * 2;
  const unsigned R_OFF  = V_OFF + 16 * 64 * 4;
  const unsigned Q_OFF  = R_OFF + 4 * 256 * 4;
  float*  S    = (float*)(smem + S_OFF);
  bf16_t* P    = (bf16_t*)(smem + P_OFF);
  float*  Vacc = (float*)(smem + V_OFF);
  float*  red  = (float*)(smem + R_OFF);
  bf16_t* Qs   = (bf16_t*)(smem + Q_OFF);

  int tid = threadIdx.x, lane = tid & 31, w = tid >> 5;
  int half = (lane >> 4) & 1, l16 = lane & 15;
  int blk = blockIdx.x;
  int rb = blk & 63;
  int bh = blk >> 6;
  int h = bh & 7, b = bh >> 3;
  size_t headoff = (size_t)(b * Hq + h);

  const bf16_t* Qh = Qb + headoff * Nq * DHq;
  const bf16_t* Kh = Kb + headoff * Nq * DHq;

  // ---- async stage of the Q tile (16 rows x 64 bf16 = 2KB) into LDS -------
  if (tid < 128) {
    const bf16_t* gp = Qh + (size_t)(rb * 16 + (tid >> 3)) * DHq + (tid & 7) * 8;
    unsigned ldsoff = __builtin_amdgcn_groupstaticsize() + Q_OFF + tid * 16;
    asm volatile("global_load_async_to_lds_b128 %0, %1, off"
                 :: "v"(ldsoff), "v"(gp)
                 : "memory");
  }
  for (int i = tid; i < 16 * 64; i += 256) Vacc[i] = 0.0f;
  asm volatile("s_wait_asynccnt 0x0" ::: "memory");
  __syncthreads();

  // ---- Phase A: score stripe S[16][SLD] = Q_tile @ K^T --------------------
  v16bf a0 = load_frag(Qs, DHq, l16, 0, half);
  v16bf a1 = load_frag(Qs, DHq, l16, 32, half);
#pragma unroll
  for (int t = 0; t < 8; ++t) {
    int col0 = w * 128 + t * 16;
    v16bf b0 = load_frag(Kh, DHq, col0 + l16, 0, half);
    v16bf b1 = load_frag(Kh, DHq, col0 + l16, 32, half);
    v8f c = {};
    c = wmma_bf16(a0, b0, c);
    c = wmma_bf16(a1, b1, c);
#pragma unroll
    for (int i = 0; i < 8; ++i)
      S[(i + half * 8) * SLD + col0 + l16] = c[i];
  }
  __syncthreads();

  // ---- Phase B: DCL sums + masked softmax ---------------------------------
  int r = tid >> 4, cl = tid & 15;         // thread -> (row, column phase)
  int grow = rb * 16 + r;
  float qn = qn_[headoff * Nq + grow];
  const float* knrow = kn_ + headoff * Nq;
  const int* mrow = mask + ((size_t)b * Nq + grow) * Nq;
  __builtin_prefetch(mrow + cl * 64, 0, 1);    // global_prefetch_b8
  __builtin_prefetch(knrow + cl * 64, 0, 2);

  float pos = 0.f, all = 0.f, mx = -3.4e38f, msum = 0.f;
  for (int c = cl; c < Nq; c += 16) {
    float s = S[r * SLD + c];
    int m = mrow[c];
    float e = __expf((s / (qn * knrow[c])) * 10.0f);   // cos/TEMP, TEMP=0.1
    all += e;
    if (m) { pos += e; msum += 1.0f; }
    float ms = (m ? s : -1.0e9f) * 0.125f;             // mask then *scale
    mx = fmaxf(mx, ms);
  }
  float* rPos = red;        float* rAll = red + 256;
  float* rMax = red + 512;  float* rMsk = red + 768;
  rPos[r * 16 + cl] = pos; rAll[r * 16 + cl] = all;
  rMax[r * 16 + cl] = mx;  rMsk[r * 16 + cl] = msum;
  __syncthreads();
  float tpos = 0.f, tall = 0.f, tmax = -3.4e38f, tmsk = 0.f;
  for (int j = 0; j < 16; ++j) {
    tpos += rPos[r * 16 + j];
    tall += rAll[r * 16 + j];
    tmax = fmaxf(tmax, rMax[r * 16 + j]);
    tmsk += rMsk[r * 16 + j];
  }
  __syncthreads();

  float sum = 0.f;
  for (int c = cl; c < Nq; c += 16) {
    float s = S[r * SLD + c];
    int m = mrow[c];
    float p = __expf((m ? s : -1.0e9f) * 0.125f - tmax);
    S[r * SLD + c] = p;
    sum += p;
  }
  rPos[r * 16 + cl] = sum;
  __syncthreads();
  float tsum = 0.f;
  for (int j = 0; j < 16; ++j) tsum += rPos[r * 16 + j];
  float inv = 1.0f / tsum;

  float* arow = attn + (headoff * Nq + grow) * Nq;
  for (int c = cl; c < Nq; c += 16) {
    float av = S[r * SLD + c] * inv;
    __builtin_nontemporal_store(av, &arow[c]);   // stream: don't pollute L2
    P[r * 1024 + c] = (bf16_t)av;
  }
  if (cl == 0) {
    atomicAdd(&accum[0], __logf(tall) - __logf(tpos));   // clustering sum
    if (h == 0) atomicAdd(&accum[1], tmsk);              // mask sum (per b once)
  }
  __syncthreads();

  // ---- Phase C: O = P @ V (V stored [b,h,dh,tok]) -------------------------
  const bf16_t* Vh = Vt + headoff * DHq * Nq;
  v8f o0 = {}, o1 = {}, o2 = {}, o3 = {};
  int kb0 = w * 128;
#pragma unroll
  for (int ks = 0; ks < 128; ks += 32) {
    v16bf pa = load_frag(P, 1024, l16, kb0 + ks, half);          // ds_read_b128
    v16bf v0 = load_frag(Vh, Nq,  0 + l16, kb0 + ks, half);
    v16bf v1 = load_frag(Vh, Nq, 16 + l16, kb0 + ks, half);
    v16bf v2 = load_frag(Vh, Nq, 32 + l16, kb0 + ks, half);
    v16bf v3 = load_frag(Vh, Nq, 48 + l16, kb0 + ks, half);
    o0 = wmma_bf16(pa, v0, o0);
    o1 = wmma_bf16(pa, v1, o1);
    o2 = wmma_bf16(pa, v2, o2);
    o3 = wmma_bf16(pa, v3, o3);
  }
#pragma unroll
  for (int t = 0; t < 4; ++t) {
    v8f o = (t == 0) ? o0 : (t == 1) ? o1 : (t == 2) ? o2 : o3;
#pragma unroll
    for (int i = 0; i < 8; ++i)
      atomicAdd(&Vacc[(i + half * 8) * 64 + t * 16 + l16], o[i]);  // ds_add_f32
  }
  __syncthreads();
  for (int i = tid; i < 16 * 64; i += 256) {
    int rr = i >> 6, dh = i & 63;
    inner[((size_t)(b * Nq + rb * 16 + rr)) * INNERq + h * DHq + dh] = (bf16_t)Vacc[i];
  }
}

// ---------------------------------------------------------------------------
// Output projection: out[4096,512] = inner_bf @ Wo + bo  (f32, NT stores)
// ---------------------------------------------------------------------------
__global__ void k_gemm_out(const bf16_t* __restrict__ A, const bf16_t* __restrict__ Wt,
                           const float* __restrict__ bias, float* __restrict__ out) {
  int tid = threadIdx.x;
  int lane = tid & 31, w = tid >> 5;
  int half = (lane >> 4) & 1, l16 = lane & 15;
  int rowblk = blockIdx.x * 64;
  int colblk = blockIdx.y * 64;
  int rowtile = (w >> 1) * 16;
  int colpair = (w & 1) * 32;

  v8f c0 = {}; v8f c1 = {};
  int arow = rowblk + rowtile + l16;
  int bcol0 = colblk + colpair;
  for (int k = 0; k < INNERq; k += 32) {
    v16bf a  = load_frag(A,  INNERq, arow,             k, half);
    v16bf b0 = load_frag(Wt, INNERq, bcol0 + l16,      k, half);
    v16bf b1 = load_frag(Wt, INNERq, bcol0 + 16 + l16, k, half);
    c0 = wmma_bf16(a, b0, c0);
    c1 = wmma_bf16(a, b1, c1);
  }
#pragma unroll
  for (int t = 0; t < 2; ++t) {
    v8f c = t ? c1 : c0;
    int col = colblk + colpair + t * 16 + l16;
    float bv = bias[col];
#pragma unroll
    for (int i = 0; i < 8; ++i) {
      int gm = rowblk + rowtile + i + half * 8;
      __builtin_nontemporal_store(c[i] + bv, &out[(size_t)gm * DIMq + col]);
    }
  }
}

// ---------------------------------------------------------------------------
// DCL finalize: mean clustering + lambda * mean regularizer
// regular_b = (sum(mask_b) - N) / (N*(N-1))  since diagonal is forced to 1
// ---------------------------------------------------------------------------
__global__ void k_dcl(const float* __restrict__ accum, float* __restrict__ out) {
  float clustering = accum[0] / (float)(Bq * Hq * Nq);
  float regular = (accum[1] - (float)(Bq * Nq)) /
                  ((float)Nq * (float)(Nq - 1) * (float)Bq);
  out[0] = clustering + 0.3f * regular;
}

// ---------------------------------------------------------------------------
extern "C" void kernel_launch(void* const* d_in, const int* in_sizes, int n_in,
                              void* d_out, int out_size, void* d_ws, size_t ws_size,
                              hipStream_t stream) {
  (void)in_sizes; (void)n_in; (void)out_size; (void)ws_size;
  const float* x  = (const float*)d_in[0];
  const int* mask = (const int*)d_in[1];
  const float* Wq = (const float*)d_in[2]; const float* bq = (const float*)d_in[3];
  const float* Wk = (const float*)d_in[4]; const float* bk = (const float*)d_in[5];
  const float* Wv = (const float*)d_in[6]; const float* bv = (const float*)d_in[7];
  const float* Wo = (const float*)d_in[8]; const float* bo = (const float*)d_in[9];

  float* out  = (float*)d_out;                         // [4,1024,512]
  float* attn = out + (size_t)Bq * Nq * DIMq;          // [4,8,1024,1024]
  float* dcl  = attn + (size_t)Bq * Hq * Nq * Nq;      // scalar

  char* ws = (char*)d_ws;
  size_t off = 0;
  auto alloc = [&](size_t bytes) -> void* {
    void* p = ws + off;
    off += (bytes + 255) & ~(size_t)255;
    return p;
  };
  bf16_t* Xb   = (bf16_t*)alloc((size_t)4096 * 512 * 2);
  bf16_t* Wqt  = (bf16_t*)alloc((size_t)512 * 512 * 2);
  bf16_t* Wkt  = (bf16_t*)alloc((size_t)512 * 512 * 2);
  bf16_t* Wvt  = (bf16_t*)alloc((size_t)512 * 512 * 2);
  bf16_t* Wot  = (bf16_t*)alloc((size_t)512 * 512 * 2);
  bf16_t* Qb   = (bf16_t*)alloc((size_t)Bq * Hq * Nq * DHq * 2);
  bf16_t* Kb   = (bf16_t*)alloc((size_t)Bq * Hq * Nq * DHq * 2);
  bf16_t* Vtb  = (bf16_t*)alloc((size_t)Bq * Hq * DHq * Nq * 2);
  bf16_t* Inb  = (bf16_t*)alloc((size_t)Bq * Nq * INNERq * 2);
  float*  qn   = (float*)alloc((size_t)Bq * Hq * Nq * 4);
  float*  kn   = (float*)alloc((size_t)Bq * Hq * Nq * 4);
  float*  accum = (float*)alloc(256);

  hipMemsetAsync(accum, 0, 2 * sizeof(float), stream);

  // converts / weight transposes
  k_cvt<<<(4096 * 512 + 255) / 256, 256, 0, stream>>>(x, Xb, 4096 * 512);
  k_cvt_t<<<(512 * 512 + 255) / 256, 256, 0, stream>>>(Wq, Wqt, 512, 512);
  k_cvt_t<<<(512 * 512 + 255) / 256, 256, 0, stream>>>(Wk, Wkt, 512, 512);
  k_cvt_t<<<(512 * 512 + 255) / 256, 256, 0, stream>>>(Wv, Wvt, 512, 512);
  k_cvt_t<<<(512 * 512 + 255) / 256, 256, 0, stream>>>(Wo, Wot, 512, 512);

  // QKV projections (+ per-row head norms for Q and K; V transposed layout)
  dim3 gg(64, 8);
  k_gemm_qkv<<<gg, 256, 0, stream>>>(Xb, Wqt, bq, Qb, qn, 0);
  k_gemm_qkv<<<gg, 256, 0, stream>>>(Xb, Wkt, bk, Kb, kn, 0);
  k_gemm_qkv<<<gg, 256, 0, stream>>>(Xb, Wvt, bv, Vtb, nullptr, 1);

  // fused scores -> dcl -> softmax -> attn write -> P@V
  size_t sh = (size_t)16 * SLD * 4 + 16 * 1024 * 2 + 16 * 64 * 4 + 4 * 256 * 4 + 2048;
  k_attn<<<Bq * Hq * (Nq / 16), 256, sh, stream>>>(Qb, Kb, Vtb, qn, kn, mask,
                                                   attn, Inb, accum);

  // output projection
  k_gemm_out<<<dim3(64, 8), 256, 0, stream>>>(Inb, Wot, bo, out);

  // scalar loss
  k_dcl<<<1, 1, 0, stream>>>(accum, dcl);
}
